// Int8LinearDynamic_16569983828459
// MI455X (gfx1250) — compile-verified, matrix operand order checked by reading, not run on hardware
//
#include <hip/hip_runtime.h>
#include <math.h>

typedef int    v8i __attribute__((ext_vector_type(8)));
typedef float  v8f __attribute__((ext_vector_type(8)));
typedef float  v2f __attribute__((ext_vector_type(2)));

#ifndef THRESHOLD_F
#define THRESHOLD_F 6.0f
#endif

static constexpr int KIN   = 2048;   // inner dim
static constexpr int KOUT  = 32;     // padded outlier column count
static constexpr int BM    = 128;    // block tile M
static constexpr int BN    = 128;    // block tile N
static constexpr int BK    = 64;     // K step (one iu8 WMMA)

#if defined(__has_builtin)
#if __has_builtin(__builtin_amdgcn_wmma_f32_16x16x4_f32)
#define HAVE_WMMA_F32X4 1
#endif
#endif
#ifndef HAVE_WMMA_F32X4
#define HAVE_WMMA_F32X4 0
#endif

// ---- CDNA5 async global->LDS copy (ASYNCcnt-tracked, GV addressing) ----
// VDST = LDS byte offset (low 32 bits of generic shared pointer == wave-relative
// LDS address; HW adds LDS_BASE), VADDR = 64-bit global address.
__device__ __forceinline__ void async_copy_b128(void* lds_ptr, const void* gptr) {
    unsigned loff = (unsigned)(unsigned long long)(size_t)lds_ptr;
    asm volatile("global_load_async_to_lds_b128 %0, %1, off"
                 :: "v"(loff), "v"(gptr)
                 : "memory");
}

__device__ __forceinline__ void wait_async0() {
#if defined(__has_builtin) && __has_builtin(__builtin_amdgcn_s_wait_asynccnt)
    __builtin_amdgcn_s_wait_asynccnt(0);
#else
    asm volatile("s_wait_asynccnt 0x0" ::: "memory");
#endif
}

// ---------------- kernel 1: zero column-max buffer ----------------
__global__ void k_init(unsigned* __restrict__ cm) {
    cm[blockIdx.x * 256 + threadIdx.x] = 0u;
}

// ---------------- kernel 2: per-column absmax (atomic exact-max => deterministic) ----------------
__global__ void k_colmax(const float* __restrict__ x, unsigned* __restrict__ cm) {
    const int col = blockIdx.x * 256 + threadIdx.x;        // gridDim.x = KIN/256
    const int r0  = blockIdx.y * 128;                      // gridDim.y = M/128
    float m = 0.0f;
    #pragma unroll 4
    for (int r = 0; r < 128; ++r)
        m = fmaxf(m, fabsf(x[(size_t)(r0 + r) * KIN + col]));
    atomicMax(&cm[col], __float_as_uint(m));               // non-negative: bit order == value order
}

// ---------------- kernel 3: parallel deterministic outlier compaction ----------------
// 1024 threads, Hillis-Steele inclusive scan over 2048 flags; idx slot = exclusive
// prefix => order independent of scheduling.
__global__ void k_scan(const unsigned* __restrict__ cm, unsigned* __restrict__ flg,
                       int* __restrict__ idx, int* __restrict__ count) {
    __shared__ int s[KIN];
    __shared__ int cnt_s;
    const int t = threadIdx.x;
    #pragma unroll
    for (int h = 0; h < 2; ++h) {
        int c = t + h * 1024;
        s[c] = (__uint_as_float(cm[c]) > THRESHOLD_F) ? 1 : 0;
    }
    __syncthreads();
    for (int off = 1; off < KIN; off <<= 1) {
        int v[2];
        #pragma unroll
        for (int h = 0; h < 2; ++h) {
            int c = t + h * 1024;
            v[h] = (c >= off) ? s[c - off] : 0;
        }
        __syncthreads();
        #pragma unroll
        for (int h = 0; h < 2; ++h)
            s[t + h * 1024] += v[h];
        __syncthreads();
    }
    #pragma unroll
    for (int h = 0; h < 2; ++h) {
        int c = t + h * 1024;
        int f = (__uint_as_float(cm[c]) > THRESHOLD_F) ? 1 : 0;
        int excl = s[c] - f;
        unsigned keep = (f && excl < KOUT) ? 1u : 0u;   // overflow cols stay on int8 path
        flg[c] = keep;
        if (keep) idx[excl] = c;
    }
    if (t == 0) cnt_s = min(s[KIN - 1], KOUT);
    __syncthreads();
    if (t < KOUT && t >= cnt_s) idx[t] = -1;
    if (t == 0) *count = cnt_s;
}

// ---------------- kernel 4: per-token activation quant + outlier gather ----------------
__global__ void k_quant_x(const float* __restrict__ x, const unsigned* __restrict__ flg,
                          const int* __restrict__ idx, signed char* __restrict__ xq,
                          float* __restrict__ sx, float* __restrict__ xo) {
    __shared__ float red[256];
    const int row = blockIdx.x;
    const int t   = threadIdx.x;
    const float* xr = x + (size_t)row * KIN;
    float m = 0.0f;
    for (int c = t; c < KIN; c += 256) {
        float v = fabsf(xr[c]);
        if (!flg[c]) m = fmaxf(m, v);
    }
    red[t] = m; __syncthreads();
    for (int s = 128; s > 0; s >>= 1) {
        if (t < s) red[t] = fmaxf(red[t], red[t + s]);
        __syncthreads();
    }
    const float scale = fmaxf(red[0] * (1.0f / 127.0f), 1e-8f);
    const float inv   = 1.0f / scale;
    for (int c = t; c < KIN; c += 256) {
        float v = flg[c] ? 0.0f : xr[c];
        float q = rintf(v * inv);                            // RNE == jnp.round
        q = fminf(fmaxf(q, -127.0f), 127.0f);
        xq[(size_t)row * KIN + c] = (signed char)q;
    }
    if (t == 0) sx[row] = scale;
    if (t < KOUT) {
        int c = idx[t];
        xo[(size_t)row * KOUT + t] = (c >= 0) ? xr[c] : 0.0f;
    }
}

// ---------------- kernel 5: per-channel weight quant + outlier gather ----------------
__global__ void k_quant_w(const float* __restrict__ w, const unsigned* __restrict__ flg,
                          const int* __restrict__ idx, signed char* __restrict__ wq,
                          float* __restrict__ sw, float* __restrict__ wo) {
    __shared__ float red[256];
    const int row = blockIdx.x;
    const int t   = threadIdx.x;
    const float* wr = w + (size_t)row * KIN;
    float m = 0.0f;
    for (int c = t; c < KIN; c += 256) {
        float v = fabsf(wr[c]);
        if (!flg[c]) m = fmaxf(m, v);
    }
    red[t] = m; __syncthreads();
    for (int s = 128; s > 0; s >>= 1) {
        if (t < s) red[t] = fmaxf(red[t], red[t + s]);
        __syncthreads();
    }
    const float scale = fmaxf(red[0] * (1.0f / 127.0f), 1e-8f);
    const float inv   = 1.0f / scale;
    for (int c = t; c < KIN; c += 256) {
        float v = flg[c] ? 0.0f : wr[c];
        float q = rintf(v * inv);
        q = fminf(fmaxf(q, -127.0f), 127.0f);
        wq[(size_t)row * KIN + c] = (signed char)q;
    }
    if (t == 0) sw[row] = scale;
    if (t < KOUT) {
        int c = idx[t];
        wo[(size_t)row * KOUT + t] = (c >= 0) ? wr[c] : 0.0f;   // original weight (fp path)
    }
}

// build 8-bit WMMA operand: per-lane 4x 8-byte K-chunks at half*8 + {0,16,32,48}
__device__ __forceinline__ v8i frag8(const signed char* rowp, int half) {
    v8i f;
    #pragma unroll
    for (int q = 0; q < 4; ++q) {
        uint2 c = *(const uint2*)(rowp + half * 8 + q * 16);
        f[2 * q]     = (int)c.x;
        f[2 * q + 1] = (int)c.y;
    }
    return f;
}

// ---------------- kernel 6: int8 WMMA GEMM, async double-buffered LDS staging,
// ----------------           fused f32-WMMA outlier epilogue ----------------
__global__ void __launch_bounds__(256)
k_gemm(const signed char* __restrict__ xq, const signed char* __restrict__ wq,
       const float* __restrict__ sx, const float* __restrict__ sw,
       const float* __restrict__ xo, const float* __restrict__ wo,
       const float* __restrict__ bias, float* __restrict__ out, int N) {
    __shared__ alignas(16) signed char sA[2][BM * BK];
    __shared__ alignas(16) signed char sB[2][BN * BK];
    __shared__ alignas(16) float sXo[BM * KOUT];
    __shared__ alignas(16) float sWo[BN * KOUT];

    const int row0 = blockIdx.y * BM;
    const int col0 = blockIdx.x * BN;
    const int t    = threadIdx.x;
    const int wid  = t >> 5;
    const int lane = t & 31;
    const int l16  = lane & 15;
    const int half = lane >> 4;
    const int wm   = wid & 3;      // 4 waves in M -> 32 rows each (2 tiles)
    const int wn   = wid >> 2;     // 2 waves in N -> 64 cols each (4 tiles)

    // issue one 128x64 int8 tile pair as async global->LDS (4x b128 per thread)
    auto issue = [&](int buf, int kb) {
        #pragma unroll
        for (int i = 0; i < 2; ++i) {
            int c = t + 256 * i;          // 512 16B chunks per tile
            int r = c >> 2, cc = (c & 3) * 16;
            async_copy_b128(&sA[buf][r * BK + cc], xq + (size_t)(row0 + r) * KIN + kb + cc);
            async_copy_b128(&sB[buf][r * BK + cc], wq + (size_t)(col0 + r) * KIN + kb + cc);
        }
    };

    issue(0, 0);                           // prologue: fill buffer 0

    // stage outlier panels (128x32 f32 each) once, via normal loads
    #pragma unroll
    for (int i = 0; i < 4; ++i) {
        int c = t + 256 * i;               // 1024 float4 chunks per panel
        int r = c >> 3, cc = (c & 7) * 4;
        *(float4*)&sXo[r * KOUT + cc] = *(const float4*)&xo[(size_t)(row0 + r) * KOUT + cc];
        *(float4*)&sWo[r * KOUT + cc] = *(const float4*)&wo[(size_t)(col0 + r) * KOUT + cc];
    }

    v8i acc[2][4];
    #pragma unroll
    for (int im = 0; im < 2; ++im)
        #pragma unroll
        for (int in = 0; in < 4; ++in)
            acc[im][in] = (v8i)0;

    wait_async0();
    __syncthreads();

    constexpr int NK = KIN / BK;
    for (int ki = 0; ki < NK; ++ki) {
        const int cur = ki & 1;
        if (ki + 1 < NK) issue(cur ^ 1, (ki + 1) * BK);   // overlap next tile with compute

        const signed char* cA = sA[cur];
        const signed char* cB = sB[cur];
        v8i af[2], bf[4];
        #pragma unroll
        for (int im = 0; im < 2; ++im)
            af[im] = frag8(&cA[(wm * 32 + im * 16 + l16) * BK], half);
        #pragma unroll
        for (int in = 0; in < 4; ++in)
            bf[in] = frag8(&cB[(wn * 64 + in * 16 + l16) * BK], half);
        #pragma unroll
        for (int im = 0; im < 2; ++im)
            #pragma unroll
            for (int in = 0; in < 4; ++in)
                acc[im][in] = __builtin_amdgcn_wmma_i32_16x16x64_iu8(
                    true, af[im], true, bf[in], acc[im][in], false, false);

        wait_async0();                     // next buffer fully landed in LDS
        __syncthreads();                   // ... and visible to all waves
    }

    // per-element scales: C layout row m = r + 8*half, col n = l16
    float sxv[2][8];
    #pragma unroll
    for (int im = 0; im < 2; ++im)
        #pragma unroll
        for (int r = 0; r < 8; ++r)
            sxv[im][r] = sx[row0 + wm * 32 + im * 16 + r + 8 * half];
    float swv[4], bv[4];
    #pragma unroll
    for (int in = 0; in < 4; ++in) {
        int n = col0 + wn * 64 + in * 16 + l16;
        swv[in] = sw[n];
        bv[in]  = bias[n];
    }

    #pragma unroll
    for (int im = 0; im < 2; ++im) {
        #pragma unroll
        for (int in = 0; in < 4; ++in) {
            v8f c;
            #pragma unroll
            for (int r = 0; r < 8; ++r)
                c[r] = (float)acc[im][in][r] * sxv[im][r] * swv[in];
#if HAVE_WMMA_F32X4
            const int mloc = wm * 32 + im * 16 + l16;   // A frag: M = lane%16
            const int nloc = wn * 64 + in * 16 + l16;   // B frag: N = lane%16
            #pragma unroll
            for (int s = 0; s < KOUT / 4; ++s) {
                v2f afp, bfp;
                afp[0] = sXo[mloc * KOUT + s * 4 + 2 * half];
                afp[1] = sXo[mloc * KOUT + s * 4 + 2 * half + 1];
                bfp[0] = sWo[nloc * KOUT + s * 4 + 2 * half];
                bfp[1] = sWo[nloc * KOUT + s * 4 + 2 * half + 1];
                c = __builtin_amdgcn_wmma_f32_16x16x4_f32(
                    false, afp, false, bfp, (short)0, c, false, false);
            }
#else
            const int nloc = wn * 64 + in * 16 + l16;
            #pragma unroll
            for (int r = 0; r < 8; ++r) {
                int mloc = wm * 32 + im * 16 + r + 8 * half;
                float a = c[r];
                for (int j = 0; j < KOUT; ++j)
                    a = fmaf(sXo[mloc * KOUT + j], sWo[nloc * KOUT + j], a);
                c[r] = a;
            }
#endif
            const int ng = col0 + wn * 64 + in * 16 + l16;
            #pragma unroll
            for (int r = 0; r < 8; ++r) {
                int mg = row0 + wm * 32 + im * 16 + r + 8 * half;
                // streamed output: keep quantized operands resident in L2
                __builtin_nontemporal_store(c[r] + bv[in], &out[(size_t)mg * N + ng]);
            }
        }
    }
}

extern "C" void kernel_launch(void* const* d_in, const int* in_sizes, int n_in,
                              void* d_out, int out_size, void* d_ws, size_t ws_size,
                              hipStream_t stream) {
    const float* x    = (const float*)d_in[0];
    const float* w    = (const float*)d_in[1];
    const float* bias = (const float*)d_in[2];
    float* out = (float*)d_out;

    const int M = in_sizes[0] / KIN;   // 8192 tokens
    const int N = in_sizes[2];         // 8192 output channels

    char* ws = (char*)d_ws;
    size_t off = 0;
    auto take = [&](size_t bytes) {
        size_t o = off;
        off += (bytes + 255) & ~(size_t)255;
        return o;
    };
    unsigned*    cm  = (unsigned*)(ws + take((size_t)KIN * 4));
    unsigned*    flg = (unsigned*)(ws + take((size_t)KIN * 4));
    int*         idx = (int*)(ws + take(64 * sizeof(int)));       // idx[0..31], count at idx[32]
    float*       sx  = (float*)(ws + take((size_t)M * 4));
    float*       sw  = (float*)(ws + take((size_t)N * 4));
    float*       xo  = (float*)(ws + take((size_t)M * KOUT * 4));
    float*       wo  = (float*)(ws + take((size_t)N * KOUT * 4));
    signed char* xq  = (signed char*)(ws + take((size_t)M * KIN));
    signed char* wq  = (signed char*)(ws + take((size_t)N * KIN));

    k_init   <<<KIN / 256, 256, 0, stream>>>(cm);
    k_colmax <<<dim3(KIN / 256, M / 128), 256, 0, stream>>>(x, cm);
    k_scan   <<<1, 1024, 0, stream>>>(cm, flg, idx, idx + 32);
    k_quant_x<<<M, 256, 0, stream>>>(x, flg, idx, xq, sx, xo);
    k_quant_w<<<N, 256, 0, stream>>>(w, flg, idx, wq, sw, wo);
    k_gemm   <<<dim3(N / BN, M / BM), 256, 0, stream>>>(xq, wq, sx, sw, xo, wo, bias, out, N);
}
